// StageNetLayer_86663850099148
// MI455X (gfx1250) — matrix operand order repaired and checked by reading, not compiled, not run on hardware
//
#include <hip/hip_runtime.h>
#include <math.h>

typedef __attribute__((ext_vector_type(16))) _Float16 v16h;
typedef __attribute__((ext_vector_type(8)))  float    v8f;

#define B_    128
#define T_    512
#define D_    128
#define H_    384
#define G_    1542
#define GP_   1552          // G padded to multiple of 16
#define CH_   128
#define CONV_ 10
#define HS_   64
#define KK_   3840          // H*CONV
#define FR_   (T_ + CONV_ - 1)   // 521 h/dist frames (9 zero frames + 512)
#define BH_   (B_ * H_)          // 49152
#define K2_   512                // D + H (bias folded out)

union U16 { v16h v; unsigned u[8]; _Float16 h[16]; };

// ---------------------------------------------------------------- prep ------
__global__ __launch_bounds__(256) void prep_kernel(
    const float* __restrict__ Wk, const float* __restrict__ bk,
    const float* __restrict__ Wr, const float* __restrict__ br,
    const float* __restrict__ Wc,
    _Float16* __restrict__ Wg, float* __restrict__ bias,
    _Float16* __restrict__ Bc, _Float16* __restrict__ hh,
    float* __restrict__ dist, float* __restrict__ cbuf)
{
  const long n1 = (long)GP_ * K2_;
  const long n2 = GP_;
  const long n3 = (long)H_ * KK_;
  const long n4 = (long)(CONV_ - 1) * BH_;
  const long n5 = (long)(CONV_ - 1) * B_;
  const long n6 = BH_;
  const long total = n1 + n2 + n3 + n4 + n5 + n6;
  for (long i = (long)blockIdx.x * 256 + threadIdx.x; i < total;
       i += (long)gridDim.x * 256) {
    long j = i;
    if (j < n1) {                       // Wg[g][k] f16, g-major, zero padded
      long g = j / K2_, k = j - g * K2_;
      float v = 0.f;
      if (g < G_) v = (k < D_) ? Wk[k * G_ + g] : Wr[(k - D_) * G_ + g];
      Wg[g * K2_ + k] = (_Float16)v;
      continue;
    }
    j -= n1;
    if (j < n2) {                       // folded bias: bk+br+Wk[last]+Wr[last]
      float v = 0.f;
      if (j < G_) v = bk[j] + br[j] + Wk[(long)D_ * G_ + j] + Wr[(long)H_ * G_ + j];
      bias[j] = v;
      continue;
    }
    j -= n2;
    if (j < n3) { Bc[j] = (_Float16)Wc[j]; continue; }   // Wc flat == [o][i*10+k]
    j -= n3;
    if (j < n4) { hh[j] = (_Float16)0.f; continue; }     // 9 zero h frames
    j -= n4;
    if (j < n5) { dist[j] = 0.f; continue; }             // 9 zero dist frames
    j -= n5;
    cbuf[j] = 0.f;                                       // c0 = 0
  }
}

// --------------------------------------------------- per-step gate GEMM ----
// x_out[128,1542] = [x_t | h_{t-1}] (f16) @ Wg^T + bias ; WMMA f16->f32
// Two independent accumulators (even/odd K chunks) so back-to-back WMMAs
// never have a D->C RAW hazard (avoids the 5-slot f16 WMMA gap).
__global__ __launch_bounds__(256) void step_gemm(
    const float* __restrict__ x, const _Float16* __restrict__ Wg,
    const float* __restrict__ bias, const _Float16* __restrict__ hh,
    float* __restrict__ xo, int t)
{
  const int w    = threadIdx.x >> 5;
  const int lane = threadIdx.x & 31;
  const int hi   = (lane & 16) ? 1 : 0;
  const int g0   = blockIdx.x * 16;
  const int rowb = w * 16;
  const int bb   = rowb + (lane & 15);
  const int gg   = g0 + (lane & 15);
  const float*    xrow = x  + ((size_t)bb * T_ + t) * D_;
  const _Float16* hrow = hh + (size_t)(CONV_ - 2 + t) * BH_ + (size_t)bb * H_;
  const _Float16* wrow = Wg + (size_t)gg * K2_;

  v8f acc0 = {0.f,0.f,0.f,0.f,0.f,0.f,0.f,0.f};
  v8f acc1 = {0.f,0.f,0.f,0.f,0.f,0.f,0.f,0.f};

  // K chunks 0..3 : x part (f32 -> f16 on the fly)
  #pragma unroll
  for (int kb = 0; kb < 4; ++kb) {
    U16 A, Bv;
    #pragma unroll
    for (int j = 0; j < 8; ++j) {
      int koff = kb * 32 + hi * 16 + 2 * j;
      A.h[2 * j]     = (_Float16)xrow[koff];
      A.h[2 * j + 1] = (_Float16)xrow[koff + 1];
    }
    #pragma unroll
    for (int j = 0; j < 8; ++j) {
      int koff = kb * 32 + hi * 16 + 2 * j;
      Bv.u[j] = *reinterpret_cast<const unsigned*>(wrow + koff);
    }
    if (kb & 1)
      acc1 = __builtin_amdgcn_wmma_f32_16x16x32_f16(false, A.v, false, Bv.v,
                                                    (short)0, acc1, false, false);
    else
      acc0 = __builtin_amdgcn_wmma_f32_16x16x32_f16(false, A.v, false, Bv.v,
                                                    (short)0, acc0, false, false);
  }
  // K chunks 4..15 : h part (already f16)
  #pragma unroll
  for (int kb = 4; kb < 16; ++kb) {
    U16 A, Bv;
    #pragma unroll
    for (int j = 0; j < 8; ++j) {
      int koff = kb * 32 + hi * 16 + 2 * j - D_;
      A.u[j] = *reinterpret_cast<const unsigned*>(hrow + koff);
    }
    #pragma unroll
    for (int j = 0; j < 8; ++j) {
      int koff = kb * 32 + hi * 16 + 2 * j;
      Bv.u[j] = *reinterpret_cast<const unsigned*>(wrow + koff);
    }
    if (kb & 1)
      acc1 = __builtin_amdgcn_wmma_f32_16x16x32_f16(false, A.v, false, Bv.v,
                                                    (short)0, acc1, false, false);
    else
      acc0 = __builtin_amdgcn_wmma_f32_16x16x32_f16(false, A.v, false, Bv.v,
                                                    (short)0, acc0, false, false);
  }
  const float bsum = bias[gg];
  #pragma unroll
  for (int v = 0; v < 8; ++v) {
    int m = v + hi * 8;                 // C layout: M = v (+8 for upper lanes)
    xo[(size_t)(rowb + m) * GP_ + gg] = acc0[v] + acc1[v] + bsum;
  }
}

// ------------------------------------------------- per-step gate update ----
__global__ __launch_bounds__(256) void step_gates(
    const float* __restrict__ xo, float* __restrict__ cbuf,
    _Float16* __restrict__ hh, float* __restrict__ dist, int t)
{
  const int tid = blockIdx.x * 256 + threadIdx.x;      // < 49152 = B*H
  const int b   = tid / H_;
  const int rem = tid - b * H_;
  const int lev = rem >> 7;
  const float* xr = xo + (size_t)b * GP_;

  // master gates (cumax l2r / r2l over 3 levels)
  float z0 = xr[0], z1 = xr[1], z2 = xr[2];
  float m1 = fmaxf(z0, fmaxf(z1, z2));
  float e0 = expf(z0 - m1), e1 = expf(z1 - m1), e2 = expf(z2 - m1);
  float inv1 = 1.f / (e0 + e1 + e2);
  float p0 = e0 * inv1, p1 = e1 * inv1;
  float fm = (lev == 0) ? p0 : ((lev == 1) ? p0 + p1 : 1.f);
  float y0 = xr[3], y1 = xr[4], y2 = xr[5];
  float m2 = fmaxf(y0, fmaxf(y1, y2));
  float q0 = expf(y0 - m2), q1 = expf(y1 - m2), q2 = expf(y2 - m2);
  float inv2 = 1.f / (q0 + q1 + q2);
  float s1 = q1 * inv2, s2 = q2 * inv2;
  float im = (lev == 0) ? 1.f : ((lev == 1) ? s1 + s2 : s2);

  int gi = 6 + rem;                                    // rem = lev*128 + c
  float fg = 1.f / (1.f + expf(-xr[gi]));
  float ig = 1.f / (1.f + expf(-xr[gi + 3 * CH_]));
  float og = 1.f / (1.f + expf(-xr[gi + 6 * CH_]));
  float ci = tanhf(xr[gi + 9 * CH_]);
  float co = cbuf[tid];
  float ov = fm * im;
  float cn = ov * (fg * co + ig * ci) + (fm - ov) * co + (im - ov) * ci;
  float hn = og * tanhf(cn);
  cbuf[tid] = cn;
  hh[(size_t)(CONV_ - 1 + t) * BH_ + tid] = (_Float16)hn;
  if (rem == 0)
    dist[(size_t)(CONV_ - 1 + t) * B_ + b] = 1.f - (p0 + (p0 + p1) + 1.f) * (1.f / 3.f);
}

// --------------------------------------- phase 3: window conv + theme ------
// block = one batch b, 16 consecutive t ; out = theme*(LH@Bc + bc) + h_t
__global__ __launch_bounds__(256) void conv_kernel(
    const _Float16* __restrict__ hh, const float* __restrict__ dist,
    const _Float16* __restrict__ Bc, const float* __restrict__ Ws,
    const float* __restrict__ bs, const float* __restrict__ Wrs,
    const float* __restrict__ brs, const float* __restrict__ bc,
    float* __restrict__ out)
{
  __shared__ _Float16 sLH[16 * 968];       // K-staged A tile (pad 960->968)
  __shared__ float    sM[16 * H_];         // mean, later theme
  __shared__ float    sHid[16 * HS_];
  __shared__ float    sDis[16 * CONV_];

  const int tid  = threadIdx.x;
  const int beta = blockIdx.x;
  const int b    = beta >> 5;
  const int t0   = (beta & 31) << 4;

  if (tid < 16) {                          // local_dis = softmax(cumsum(window))
    int r = tid;
    float cum = 0.f, vals[CONV_];
    #pragma unroll
    for (int k = 0; k < CONV_; ++k) {
      cum += dist[(size_t)(t0 + r + k) * B_ + b];
      vals[k] = cum;
    }
    float mx = vals[0];
    #pragma unroll
    for (int k = 1; k < CONV_; ++k) mx = fmaxf(mx, vals[k]);
    float s = 0.f, ev[CONV_];
    #pragma unroll
    for (int k = 0; k < CONV_; ++k) { ev[k] = expf(vals[k] - mx); s += ev[k]; }
    float invs = 1.f / s;
    #pragma unroll
    for (int k = 0; k < CONV_; ++k) sDis[r * CONV_ + k] = ev[k] * invs;
  }
  __syncthreads();

  const int w = tid >> 5, lane = tid & 31;
  const int hi = (lane & 16) ? 1 : 0;
  const int rA = lane & 15;
  const unsigned* sLHu = reinterpret_cast<const unsigned*>(sLH);
  v8f accv[3];
  #pragma unroll
  for (int tau = 0; tau < 3; ++tau)
    accv[tau] = (v8f){0.f,0.f,0.f,0.f,0.f,0.f,0.f,0.f};

  for (int pass = 0; pass < 4; ++pass) {           // K = 3840 in 4 x 960
    for (int q = 0; q < 60; ++q) {                 // stage LH (h * dis) as f16
      int e = tid + q * 256;
      int r = e / 960, l = e - r * 960;
      int il = l / 10, k = l - il * 10;
      int ii = pass * 96 + il;
      float hv = (float)hh[(size_t)(t0 + r + k) * BH_ + (size_t)b * H_ + ii];
      sLH[r * 968 + l] = (_Float16)(hv * sDis[r * CONV_ + k]);
    }
    __syncthreads();
    for (int q = 0; q < 6; ++q) {                  // mean over window
      int e = tid + q * 256;
      int r = e / 96, il = e - r * 96;
      float s = 0.f;
      #pragma unroll
      for (int k = 0; k < CONV_; ++k) s += (float)sLH[r * 968 + il * 10 + k];
      sM[r * H_ + pass * 96 + il] = s * 0.1f;
    }
    for (int ch = 0; ch < 30; ++ch) {              // WMMA over this K slab
      U16 A;
      #pragma unroll
      for (int j = 0; j < 8; ++j) {
        int l = ch * 32 + hi * 16 + 2 * j;
        A.u[j] = sLHu[(rA * 968 + l) >> 1];
      }
      int kkb = pass * 960 + ch * 32 + hi * 16;
      #pragma unroll
      for (int tau = 0; tau < 3; ++tau) {
        int o = (3 * w + tau) * 16 + (lane & 15);
        U16 Bv;
        #pragma unroll
        for (int j = 0; j < 8; ++j)
          Bv.u[j] = *reinterpret_cast<const unsigned*>(Bc + (size_t)o * KK_ + kkb + 2 * j);
        accv[tau] = __builtin_amdgcn_wmma_f32_16x16x32_f16(
            false, A.v, false, Bv.v, (short)0, accv[tau], false, false);
      }
    }
    __syncthreads();
  }

  for (int q = 0; q < 4; ++q) {                    // hid = relu(mean@Ws+bs)
    int e = tid + q * 256;
    int r = e >> 6, j = e & 63;
    float s = bs[j];
    for (int i = 0; i < H_; ++i) s += sM[r * H_ + i] * Ws[i * HS_ + j];
    sHid[e] = fmaxf(s, 0.f);
  }
  __syncthreads();
  float th[24];
  for (int q = 0; q < 24; ++q) {                   // theme = sigmoid(hid@Wrs+brs)
    int e = tid + q * 256;
    int r = e / H_, o = e - r * H_;
    float s = brs[o];
    #pragma unroll 4
    for (int j = 0; j < HS_; ++j) s += sHid[r * HS_ + j] * Wrs[j * H_ + o];
    th[q] = 1.f / (1.f + expf(-s));
  }
  __syncthreads();
  for (int q = 0; q < 24; ++q) {
    int e = tid + q * 256;
    sM[e] = th[q];                                 // overwrite mean with theme
  }
  __syncthreads();

  #pragma unroll
  for (int tau = 0; tau < 3; ++tau) {              // out = theme*(conv+bc)+h_t
    int o = (3 * w + tau) * 16 + (lane & 15);
    float bcv = bc[o];
    #pragma unroll
    for (int v = 0; v < 8; ++v) {
      int r = v + hi * 8;
      int t = t0 + r;
      float hv = (float)hh[(size_t)(CONV_ - 1 + t) * BH_ + (size_t)b * H_ + o];
      out[((size_t)b * T_ + t) * H_ + o] = sM[r * H_ + o] * (accv[tau][v] + bcv) + hv;
    }
  }
}

// ---------------------------------------------------------------- launch ---
extern "C" void kernel_launch(void* const* d_in, const int* in_sizes, int n_in,
                              void* d_out, int out_size, void* d_ws, size_t ws_size,
                              hipStream_t stream) {
  (void)in_sizes; (void)n_in; (void)out_size; (void)ws_size;
  const float* x   = (const float*)d_in[0];
  const float* Wk  = (const float*)d_in[2];
  const float* bk  = (const float*)d_in[3];
  const float* Wr  = (const float*)d_in[4];
  const float* br  = (const float*)d_in[5];
  const float* Ws  = (const float*)d_in[6];
  const float* bs  = (const float*)d_in[7];
  const float* Wrs = (const float*)d_in[8];
  const float* brs = (const float*)d_in[9];
  const float* Wc  = (const float*)d_in[10];
  const float* bc  = (const float*)d_in[11];

  char* ws = (char*)d_ws;
  const size_t oWg  = 0;                                  // f16 [GP_][512]
  const size_t oBia = oWg  + (size_t)GP_ * K2_ * 2;       // f32 [GP_]
  const size_t oBc  = oBia + (size_t)GP_ * 4;             // f16 [H_][KK_]
  const size_t oHh  = oBc  + (size_t)H_ * KK_ * 2;        // f16 [FR_][B_][H_]
  const size_t oDi  = oHh  + (size_t)FR_ * BH_ * 2;       // f32 [FR_][B_]
  const size_t oC   = oDi  + (size_t)FR_ * B_ * 4;        // f32 [B_][H_]
  const size_t oXo  = oC   + (size_t)BH_ * 4;             // f32 [B_][GP_]
  _Float16* Wg   = (_Float16*)(ws + oWg);
  float*    bias = (float*)   (ws + oBia);
  _Float16* Bcm  = (_Float16*)(ws + oBc);
  _Float16* hh   = (_Float16*)(ws + oHh);
  float*    dist = (float*)   (ws + oDi);
  float*    cbuf = (float*)   (ws + oC);
  float*    xo   = (float*)   (ws + oXo);

  prep_kernel<<<1024, 256, 0, stream>>>(Wk, bk, Wr, br, Wc,
                                        Wg, bias, Bcm, hh, dist, cbuf);
  for (int t = 0; t < T_; ++t) {
    step_gemm<<<GP_ / 16, 256, 0, stream>>>(x, Wg, bias, hh, xo, t);
    step_gates<<<BH_ / 256, 256, 0, stream>>>(xo, cbuf, hh, dist, t);
  }
  conv_kernel<<<B_ * (T_ / 16), 256, 0, stream>>>(hh, dist, Bcm, Ws, bs,
                                                  Wrs, brs, bc, (float*)d_out);
}
